// MultiHeadAttentionBlock_51092930953805
// MI455X (gfx1250) — matrix operand order repaired and checked by reading, not compiled, not run on hardware
//
#include <hip/hip_runtime.h>

// ---------------------------------------------------------------------------
// MHA block for MI455X (gfx1250, wave32, WMMA bf16 16x16x32 + async-to-LDS)
// B=2, S=2048, D=1024, H=16, DK=64
// ---------------------------------------------------------------------------

#define BATCH 2
#define SEQ   2048
#define DMODEL 1024
#define NHEAD 16
#define DHEAD 64
#define MROWS (BATCH * SEQ)   // 4096

typedef __attribute__((ext_vector_type(16))) __bf16 v16bf;
typedef __attribute__((ext_vector_type(8)))  __bf16 v8bf;
typedef __attribute__((ext_vector_type(8)))  float  v8f;
typedef __attribute__((ext_vector_type(4)))  float  v4f;

static __device__ inline v8f wmma_bf16(v16bf a, v16bf b, v8f c) {
    // D = A(16x32 bf16) * B(32x16 bf16) + C(16x16 f32)
    return __builtin_amdgcn_wmma_f32_16x16x32_bf16(
        /*neg_a=*/false, a, /*neg_b=*/false, b,
        /*c_mod=*/(short)0, c, /*reuse_a=*/false, /*reuse_b=*/false);
}

static __device__ inline v16bf cat8(v8bf lo, v8bf hi) {
    v16bf r;
#pragma unroll
    for (int i = 0; i < 8; ++i) { r[i] = lo[i]; r[i + 8] = hi[i]; }
    return r;
}

// A-fragment (16x32, 16-bit): lane row = lane&15; per-lane K runs are
// [g*8 .. g*8+7] and [16+g*8 .. 16+g*8+7]  (g = lane>>4).
static __device__ inline v16bf afrag_from_f32(const float* ap) {
    v4f a0 = *(const v4f*)(ap + 0);
    v4f a1 = *(const v4f*)(ap + 4);
    v4f a2 = *(const v4f*)(ap + 16);
    v4f a3 = *(const v4f*)(ap + 20);
    v16bf r;
#pragma unroll
    for (int i = 0; i < 4; ++i) {
        r[i]      = (__bf16)a0[i];
        r[4 + i]  = (__bf16)a1[i];
        r[8 + i]  = (__bf16)a2[i];
        r[12 + i] = (__bf16)a3[i];
    }
    return r;
}

// B-fragment (32x16, 16-bit): lane col = lane&15; per-lane K runs are
// [g*8 .. g*8+7] and [16+g*8 .. 16+g*8+7]: two 16B loads from bf16 memory.
static __device__ inline v16bf bfrag_from_bf16(const __bf16* bp) {
    return cat8(*(const v8bf*)bp, *(const v8bf*)(bp + 16));
}

static __device__ inline float red_max16(float v) {
    v = fmaxf(v, __shfl_xor(v, 1));
    v = fmaxf(v, __shfl_xor(v, 2));
    v = fmaxf(v, __shfl_xor(v, 4));
    v = fmaxf(v, __shfl_xor(v, 8));
    return v;
}
static __device__ inline float red_sum16(float v) {
    v += __shfl_xor(v, 1);
    v += __shfl_xor(v, 2);
    v += __shfl_xor(v, 4);
    v += __shfl_xor(v, 8);
    return v;
}

// CDNA5 async copy: 16B global -> LDS, tracked with ASYNCcnt (no VGPR round trip)
static __device__ inline void async_copy16(unsigned int lds_off, const void* gptr) {
    asm volatile("global_load_async_to_lds_b128 %0, %1, off"
                 :: "v"(lds_off), "v"((unsigned long long)(uintptr_t)gptr)
                 : "memory");
}
static __device__ inline void wait_async0() {
    asm volatile("s_wait_asynccnt 0x0" ::: "memory");
}

// ---------------------------------------------------------------------------
// Kernel 0: one-time fp32 -> bf16 conversion of the 4 weight matrices.
//  grid = (512, 4), block = 256; 8 elements/thread.
// ---------------------------------------------------------------------------
__global__ __launch_bounds__(256)
void wcvt_kernel(const float* __restrict__ Wq, const float* __restrict__ Wk,
                 const float* __restrict__ Wv, const float* __restrict__ Wo,
                 __bf16* __restrict__ dst) {
    const int mtx = blockIdx.y;
    const float* src = (mtx == 0) ? Wq : (mtx == 1) ? Wk : (mtx == 2) ? Wv : Wo;
    __bf16* d = dst + (size_t)mtx * DMODEL * DMODEL;
    const size_t base = (size_t)(blockIdx.x * 256 + threadIdx.x) * 8;
    v4f a = *(const v4f*)(src + base);
    v4f b = *(const v4f*)(src + base + 4);
    v8bf o;
#pragma unroll
    for (int i = 0; i < 4; ++i) { o[i] = (__bf16)a[i]; o[4 + i] = (__bf16)b[i]; }
    *(v8bf*)(d + base) = o;
}

// ---------------------------------------------------------------------------
// Kernel 1: fused QKV projection.  grid = (M/64, D/64, 3), block = 128 (4 waves)
//  p==0: Q -> Qh[b,h,s,dk] bf16, pre-scaled by 1/sqrt(DK)
//  p==1: K -> Kh[b,h,s,dk] bf16
//  p==2: V -> Vt[b,h,dk,s] bf16 (transposed for the PV GEMM B-layout)
// ---------------------------------------------------------------------------
__global__ __launch_bounds__(128)
void qkv_proj_kernel(const float* __restrict__ q, const float* __restrict__ k,
                     const float* __restrict__ v, const __bf16* __restrict__ Wb,
                     const float* __restrict__ bq, const float* __restrict__ bk,
                     const float* __restrict__ bv,
                     __bf16* __restrict__ Qh, __bf16* __restrict__ Kh,
                     __bf16* __restrict__ Vt) {
    const int tid  = threadIdx.x;
    const int wave = tid >> 5;
    const int lane = tid & 31;
    const int col  = lane & 15;
    const int grp  = lane >> 4;
    const int p    = blockIdx.z;

    const float*  X    = (p == 0) ? q  : (p == 1) ? k  : v;
    const __bf16* W    = Wb + (size_t)p * DMODEL * DMODEL;
    const float*  bias = (p == 0) ? bq : (p == 1) ? bk : bv;

    const int m0 = blockIdx.x * 64 + wave * 16;
    const int n0 = blockIdx.y * 64;
    const int arow = m0 + col;

    v8f acc[4] = {{}, {}, {}, {}};

    for (int kk = 0; kk < DMODEL; kk += 32) {
        v16bf a = afrag_from_f32(X + (size_t)arow * DMODEL + kk + grp * 8);
#pragma unroll
        for (int f = 0; f < 4; ++f) {
            const int n = n0 + f * 16 + col;
            v16bf b = bfrag_from_bf16(W + (size_t)n * DMODEL + kk + grp * 8);
            acc[f] = wmma_bf16(a, b, acc[f]);
        }
    }

#pragma unroll
    for (int f = 0; f < 4; ++f) {
        const int n  = n0 + f * 16 + col;
        const int hh = n >> 6;     // head
        const int dk = n & 63;     // dim within head
        const float bb = bias[n];
#pragma unroll
        for (int r = 0; r < 8; ++r) {
            const int m  = m0 + r + 8 * grp;   // C-layout row
            const int bi = m >> 11;            // batch
            const int s  = m & (SEQ - 1);      // seq pos
            float val = acc[f][r] + bb;
            const size_t bh = (size_t)(bi * NHEAD + hh);
            if (p == 0) {
                Qh[(bh * SEQ + s) * DHEAD + dk] = (__bf16)(val * 0.125f);
            } else if (p == 1) {
                Kh[(bh * SEQ + s) * DHEAD + dk] = (__bf16)val;
            } else {
                Vt[(bh * DHEAD + dk) * SEQ + s] = (__bf16)val;
            }
        }
    }
}

// ---------------------------------------------------------------------------
// Kernel 2: causal flash attention.  grid = (S/64, B*H), block = 128 (4 waves)
//  K/V tiles cooperatively staged in LDS via global_load_async_to_lds_b128;
//  all waves iterate the block-uniform key range (barrier-safe), lagging waves
//  skip compute only.
// ---------------------------------------------------------------------------
__global__ __launch_bounds__(128)
void attn_kernel(const __bf16* __restrict__ Qh, const __bf16* __restrict__ Kh,
                 const __bf16* __restrict__ Vt, __bf16* __restrict__ Ctx) {
    __shared__ __align__(16) __bf16 Klds[32 * DHEAD];     // 32 keys x 64 dk (4KB)
    __shared__ __align__(16) __bf16 Vlds[DHEAD * 32];     // 64 dims x 32 keys (4KB)
    __shared__ __align__(16) __bf16 plds[4][16 * 32];     // per-wave P transpose

    const int tid  = threadIdx.x;
    const int wave = tid >> 5;
    const int lane = tid & 31;
    const int col  = lane & 15;
    const int grp  = lane >> 4;
    const int bh   = blockIdx.y;
    const int bi   = bh >> 4;
    const int hh   = bh & 15;
    const int q0b  = blockIdx.x * 64;
    const int q0   = q0b + wave * 16;

    const __bf16* Qb = Qh + (size_t)bh * SEQ * DHEAD;
    const __bf16* Kb = Kh + (size_t)bh * SEQ * DHEAD;
    const __bf16* Vb = Vt + (size_t)bh * DHEAD * SEQ;
    __bf16* pw = &plds[wave][0];

    // Q A-fragments held in registers for the whole key loop
    v16bf qa[2];
#pragma unroll
    for (int t = 0; t < 2; ++t) {
        const __bf16* qp = Qb + (size_t)(q0 + col) * DHEAD + t * 32 + grp * 8;
        qa[t] = cat8(*(const v8bf*)qp, *(const v8bf*)(qp + 16));
    }

    v8f cacc[4] = {{}, {}, {}, {}};
    float mrow[8], lrow[8];
#pragma unroll
    for (int r = 0; r < 8; ++r) { mrow[r] = -1e30f; lrow[r] = 0.0f; }

    const int kbEnd    = (q0 + 15) >> 5;        // this wave's last key block
    const int kbEndBlk = (q0b + 63) >> 5;       // block-uniform loop bound

    for (int kb = 0; kb <= kbEndBlk; ++kb) {
        const int k0 = kb << 5;

        // ---- cooperative async staging: K 32x64, V 64x32 (bf16) ----
        // K: 256 16B chunks; 8 chunks per key row
#pragma unroll
        for (int i = 0; i < 2; ++i) {
            const int c   = tid + i * 128;
            const int row = c >> 3;
            const int off = (c & 7) * 8;
            async_copy16((unsigned int)(uintptr_t)&Klds[row * DHEAD + off],
                         Kb + (size_t)(k0 + row) * DHEAD + off);
        }
        // V: 256 16B chunks; 4 chunks per dim row
#pragma unroll
        for (int i = 0; i < 2; ++i) {
            const int c   = tid + i * 128;
            const int row = c >> 2;
            const int off = (c & 3) * 8;
            async_copy16((unsigned int)(uintptr_t)&Vlds[row * 32 + off],
                         Vb + (size_t)row * SEQ + k0 + off);
        }
        wait_async0();
        __syncthreads();

        if (kb <= kbEnd) {
            // ---- scores: S = (Q/sqrt(dk)) @ K^T, 16x32 via 2x2 WMMAs ----
            v8f s0 = {}, s1 = {};
#pragma unroll
            for (int t = 0; t < 2; ++t) {
                v16bf b0 = bfrag_from_bf16(&Klds[(col)      * DHEAD + t * 32 + grp * 8]);
                v16bf b1 = bfrag_from_bf16(&Klds[(16 + col) * DHEAD + t * 32 + grp * 8]);
                s0 = wmma_bf16(qa[t], b0, s0);
                s1 = wmma_bf16(qa[t], b1, s1);
            }

            // ---- causal mask (only diagonal-overlapping blocks) ----
            if (k0 + 31 > q0) {
#pragma unroll
                for (int r = 0; r < 8; ++r) {
                    const int qi = q0 + r + 8 * grp;
                    if (k0 + col > qi)      s0[r] = -1e9f;
                    if (k0 + 16 + col > qi) s1[r] = -1e9f;
                }
            }

            // ---- online softmax ----
            float alpha[8];
#pragma unroll
            for (int r = 0; r < 8; ++r) {
                float mb = red_max16(fmaxf(s0[r], s1[r]));
                float mn = fmaxf(mrow[r], mb);
                alpha[r] = __expf(mrow[r] - mn);
                float p0 = __expf(s0[r] - mn);
                float p1 = __expf(s1[r] - mn);
                s0[r] = p0; s1[r] = p1;
                lrow[r] = lrow[r] * alpha[r] + red_sum16(p0 + p1);
                mrow[r] = mn;
            }
#pragma unroll
            for (int r = 0; r < 8; ++r) {
                cacc[0][r] *= alpha[r]; cacc[1][r] *= alpha[r];
                cacc[2][r] *= alpha[r]; cacc[3][r] *= alpha[r];
            }

            // ---- P: C-layout -> A-layout through wave-private LDS ----
            __builtin_amdgcn_wave_barrier();
#pragma unroll
            for (int r = 0; r < 8; ++r) {
                pw[(r + 8 * grp) * 32 + col]      = (__bf16)s0[r];
                pw[(r + 8 * grp) * 32 + 16 + col] = (__bf16)s1[r];
            }
            __builtin_amdgcn_wave_barrier();
            const __bf16* pp = pw + col * 32 + grp * 8;
            v16bf pa = cat8(*(const v8bf*)pp, *(const v8bf*)(pp + 16));
            __builtin_amdgcn_wave_barrier();

            // ---- ctx += P @ V  (V tile rows = dims, contiguous key runs) ----
#pragma unroll
            for (int f = 0; f < 4; ++f) {
                v16bf vb = bfrag_from_bf16(&Vlds[(f * 16 + col) * 32 + grp * 8]);
                cacc[f] = wmma_bf16(pa, vb, cacc[f]);
            }
        }
        __syncthreads();   // protect K/V tiles before next iteration's staging
    }

    // ---- normalize and write ctx as [B, S, D] bf16 ----
#pragma unroll
    for (int r = 0; r < 8; ++r) {
        const int si = q0 + r + 8 * grp;
        const float inv = 1.0f / lrow[r];
        const size_t base = ((size_t)(bi * SEQ + si)) * DMODEL + hh * DHEAD + col;
        Ctx[base + 0]  = (__bf16)(cacc[0][r] * inv);
        Ctx[base + 16] = (__bf16)(cacc[1][r] * inv);
        Ctx[base + 32] = (__bf16)(cacc[2][r] * inv);
        Ctx[base + 48] = (__bf16)(cacc[3][r] * inv);
    }
}

// ---------------------------------------------------------------------------
// Kernel 3: output projection  out = Ctx @ Wo^T + bo  (fp32 out)
// ---------------------------------------------------------------------------
__global__ __launch_bounds__(128)
void out_proj_kernel(const __bf16* __restrict__ Ctx, const __bf16* __restrict__ Wob,
                     const float* __restrict__ bo, float* __restrict__ out) {
    const int tid  = threadIdx.x;
    const int wave = tid >> 5;
    const int lane = tid & 31;
    const int col  = lane & 15;
    const int grp  = lane >> 4;

    const int m0 = blockIdx.x * 64 + wave * 16;
    const int n0 = blockIdx.y * 64;
    const int arow = m0 + col;

    v8f acc[4] = {{}, {}, {}, {}};

    for (int kk = 0; kk < DMODEL; kk += 32) {
        const __bf16* ap = Ctx + (size_t)arow * DMODEL + kk + grp * 8;
        v16bf a = cat8(*(const v8bf*)ap, *(const v8bf*)(ap + 16));
#pragma unroll
        for (int f = 0; f < 4; ++f) {
            const int n = n0 + f * 16 + col;
            v16bf b = bfrag_from_bf16(Wob + (size_t)n * DMODEL + kk + grp * 8);
            acc[f] = wmma_bf16(a, b, acc[f]);
        }
    }

#pragma unroll
    for (int f = 0; f < 4; ++f) {
        const int n = n0 + f * 16 + col;
        const float bb = bo[n];
#pragma unroll
        for (int r = 0; r < 8; ++r) {
            const int m = m0 + r + 8 * grp;
            out[(size_t)m * DMODEL + n] = acc[f][r] + bb;
        }
    }
}

// ---------------------------------------------------------------------------
extern "C" void kernel_launch(void* const* d_in, const int* in_sizes, int n_in,
                              void* d_out, int out_size, void* d_ws, size_t ws_size,
                              hipStream_t stream) {
    const float* q  = (const float*)d_in[0];
    const float* k  = (const float*)d_in[1];
    const float* v  = (const float*)d_in[2];
    // d_in[3] = causal mask (implemented analytically in-kernel)
    const float* Wq = (const float*)d_in[4];
    const float* bq = (const float*)d_in[5];
    const float* Wk = (const float*)d_in[6];
    const float* bk = (const float*)d_in[7];
    const float* Wv = (const float*)d_in[8];
    const float* bv = (const float*)d_in[9];
    const float* Wo = (const float*)d_in[10];
    const float* bo = (const float*)d_in[11];
    float* out = (float*)d_out;

    const size_t per  = (size_t)BATCH * NHEAD * SEQ * DHEAD;  // 4M bf16 elems
    const size_t wsz  = (size_t)DMODEL * DMODEL;              // 1M elems per W
    __bf16* Qh  = (__bf16*)d_ws;
    __bf16* Kh  = Qh + per;
    __bf16* Vt  = Kh + per;
    __bf16* Ctx = Vt + per;        // [B, S, D] bf16
    __bf16* Wb  = Ctx + per;       // Wq,Wk,Wv,Wo bf16 — total ws use: 40 MB

    wcvt_kernel<<<dim3(512, 4), 256, 0, stream>>>(Wq, Wk, Wv, Wo, Wb);

    qkv_proj_kernel<<<dim3(MROWS / 64, DMODEL / 64, 3), 128, 0, stream>>>(
        q, k, v, Wb, bq, bk, bv, Qh, Kh, Vt);

    attn_kernel<<<dim3(SEQ / 64, BATCH * NHEAD), 128, 0, stream>>>(Qh, Kh, Vt, Ctx);

    out_proj_kernel<<<dim3(MROWS / 64, DMODEL / 64), 128, 0, stream>>>(
        Ctx, Wb + 3 * wsz, bo, out);
}
